// Network_13168369729592
// MI455X (gfx1250) — compile-verified, matrix operand order checked by reading, not run on hardware
//
#include <hip/hip_runtime.h>

// ---------------------------------------------------------------------------
// CDNA5 (gfx1250) fused GraphSAGE forward, bf16 WMMA (f32 accum).
//   - all GEMM operands pre-converted to bf16 (no cvt in the K-loop)
//   - edge gather reads bf16 features (halves dominant random traffic)
//   - one wave computes a 32x32 output tile: 8 wmma per K=32 step
//   - layer-2 epilogue fuses the graph mean-pool (no h2 buffer at all)
// ---------------------------------------------------------------------------

typedef __attribute__((ext_vector_type(16))) __bf16 v16bf;
typedef __attribute__((ext_vector_type(8)))  __bf16 v8bf;
typedef __attribute__((ext_vector_type(8)))  float  v8f;

// Fragment loader: 16 bf16 elements for this lane at K-slab k0.
// ISA 16-bit A/B layout: VGPR v -> K = (v>>2)*16 + hi*8 + (v&3)*2 (+0/1),
// i.e. two contiguous 8-element (16B) runs at k0+hi*8 and k0+16+hi*8.
__device__ __forceinline__ v16bf load_frag(const __bf16* __restrict__ row,
                                           int k0, int hi) {
  v8bf lo = *(const v8bf*)(row + k0 + hi * 8);
  v8bf hh = *(const v8bf*)(row + k0 + 16 + hi * 8);
  return __builtin_shufflevector(lo, hh, 0, 1, 2, 3, 4, 5, 6, 7,
                                         8, 9, 10, 11, 12, 13, 14, 15);
}

// ---- degree count -----------------------------------------------------------
__global__ __launch_bounds__(256) void degree_kernel(const int* __restrict__ dst,
                                                     float* __restrict__ deg, int E) {
  int i = blockIdx.x * blockDim.x + threadIdx.x;
  if (i < E) atomicAdd(&deg[dst[i]], 1.0f);
}

__global__ __launch_bounds__(256) void invdeg_kernel(float* __restrict__ deg, int n) {
  int i = blockIdx.x * blockDim.x + threadIdx.x;
  if (i < n) deg[i] = 1.0f / fmaxf(deg[i], 1.0f);
}

// ---- f32 -> bf16 linear convert (weights, x) --------------------------------
__global__ __launch_bounds__(256) void cvt_bf16_kernel(const float* __restrict__ src,
                                                       __bf16* __restrict__ dst,
                                                       long long count) {
  for (long long i = (long long)blockIdx.x * blockDim.x + threadIdx.x; i < count;
       i += (long long)gridDim.x * blockDim.x)
    dst[i] = (__bf16)src[i];
}

// ---- agg(f32) * invdeg -> bf16 ---------------------------------------------
__global__ __launch_bounds__(256) void scale_cvt_kernel(const float* __restrict__ agg,
                                                        const float* __restrict__ invdeg,
                                                        __bf16* __restrict__ dst,
                                                        int n, int K) {
  const long long total = (long long)n * K;
  for (long long i = (long long)blockIdx.x * blockDim.x + threadIdx.x; i < total;
       i += (long long)gridDim.x * blockDim.x) {
    const int row = (int)(i / K);
    dst[i] = (__bf16)(agg[i] * invdeg[row]);
  }
}

// ---- edge-parallel scatter-add: bf16 features -> f32 accumulators -----------
// One thread per (edge, 8-feature chunk): one 16B vector read, 8 f32 atomics.
__global__ __launch_bounds__(256) void scatter_add_kernel(const __bf16* __restrict__ featb,
                                                          const int* __restrict__ src,
                                                          const int* __restrict__ dst,
                                                          float* __restrict__ agg,
                                                          int E, int K) {
  const int chunks = K >> 3;
  const long long total = (long long)E * chunks;
  for (long long i = (long long)blockIdx.x * blockDim.x + threadIdx.x; i < total;
       i += (long long)gridDim.x * blockDim.x) {
    const int e = (int)(i / chunks);
    const int c = ((int)(i - (long long)e * chunks)) << 3;
    const size_t s = (size_t)src[e];
    const size_t d = (size_t)dst[e];
    const v8bf v = *(const v8bf*)(featb + s * K + c);
    float* o = agg + d * K + c;
#pragma unroll
    for (int j = 0; j < 8; ++j) atomicAdd(o + j, (float)v[j]);
  }
}

// ---- per-tile epilogue: leaky_relu, optional bf16 store, optional pool ------
__device__ __forceinline__ void emit_tile(const v8f& acc, int rowbase, int col,
                                          int hi, int n_nodes, int O,
                                          __bf16* __restrict__ houtb,
                                          float* __restrict__ gsum) {
  float s = 0.0f;
#pragma unroll
  for (int r = 0; r < 8; ++r) {
    const int row = rowbase + r + 8 * hi;
    float v = acc[r];
    v = v > 0.0f ? v : 0.01f * v;
    if (houtb && row < n_nodes) houtb[(size_t)row * O + col] = (__bf16)v;
    if (gsum) s += (row < n_nodes) ? v : 0.0f;
  }
  if (gsum) atomicAdd(&gsum[col], s);
}

// ---- fused SAGE layer: one wave = one 32x32 output super-tile ---------------
// D = leaky_relu( gb @ Wl^T + bias + hin @ Wr^T ),  gb = bf16(agg/deg)
__global__ __launch_bounds__(256) void sage_wmma_kernel(
    const __bf16* __restrict__ hinb,  // [n,K] root features (bf16)
    const __bf16* __restrict__ gb,    // [n,K] scaled aggregate (bf16)
    const __bf16* __restrict__ Wlb,   // [O,K]
    const __bf16* __restrict__ Wrb,   // [O,K]
    const float*  __restrict__ bias,  // [O]
    __bf16* __restrict__ houtb,       // [n,O] bf16 or nullptr
    float*  __restrict__ gsum,        // [O] pooled accumulator or nullptr
    int n_nodes, int K, int O)
{
  const int lane = threadIdx.x & 31;
  const int wave = blockIdx.x * (blockDim.x >> 5) + (threadIdx.x >> 5);
  const int otiles = O >> 5;                       // 32-wide super-tiles
  const int ntiles = (n_nodes + 31) >> 5;
  if (wave >= ntiles * otiles) return;             // wave-uniform: EXEC stays full
  const int nb = (wave / otiles) << 5;
  const int ob = (wave % otiles) << 5;

  const int m  = lane & 15;
  const int hi = lane >> 4;

  int n0 = nb + m;       if (n0 >= n_nodes) n0 = n_nodes - 1;   // clamped; masked later
  int n1 = nb + 16 + m;  if (n1 >= n_nodes) n1 = n_nodes - 1;

  const __bf16* __restrict__ a0 = hinb + (size_t)n0 * K;
  const __bf16* __restrict__ a1 = hinb + (size_t)n1 * K;
  const __bf16* __restrict__ g0 = gb   + (size_t)n0 * K;
  const __bf16* __restrict__ g1 = gb   + (size_t)n1 * K;
  const __bf16* __restrict__ l0 = Wlb + (size_t)(ob + m) * K;
  const __bf16* __restrict__ l1 = Wlb + (size_t)(ob + 16 + m) * K;
  const __bf16* __restrict__ r0 = Wrb + (size_t)(ob + m) * K;
  const __bf16* __restrict__ r1 = Wrb + (size_t)(ob + 16 + m) * K;

  // C/D layout: VGPR r, lane L -> row base + r + 8*(L>=16), col base + (L&15).
  const float bv0 = bias[ob + m];
  const float bv1 = bias[ob + 16 + m];
  v8f acc00, acc01, acc10, acc11;
#pragma unroll
  for (int r = 0; r < 8; ++r) {
    acc00[r] = bv0; acc10[r] = bv0;
    acc01[r] = bv1; acc11[r] = bv1;
  }

  for (int k0 = 0; k0 < K; k0 += 32) {
    if (k0 + 32 < K) {                             // prefetch streaming A operands
      __builtin_prefetch(a0 + k0 + 32, 0, 3);
      __builtin_prefetch(a1 + k0 + 32, 0, 3);
      __builtin_prefetch(g0 + k0 + 32, 0, 3);
      __builtin_prefetch(g1 + k0 + 32, 0, 3);
    }
    const v16bf A0 = load_frag(a0, k0, hi);
    const v16bf A1 = load_frag(a1, k0, hi);
    const v16bf G0 = load_frag(g0, k0, hi);
    const v16bf G1 = load_frag(g1, k0, hi);
    const v16bf L0 = load_frag(l0, k0, hi);
    const v16bf L1 = load_frag(l1, k0, hi);
    const v16bf R0 = load_frag(r0, k0, hi);
    const v16bf R1 = load_frag(r1, k0, hi);

    acc00 = __builtin_amdgcn_wmma_f32_16x16x32_bf16(false, G0, false, L0, (short)0, acc00, false, false);
    acc00 = __builtin_amdgcn_wmma_f32_16x16x32_bf16(false, A0, false, R0, (short)0, acc00, false, false);
    acc01 = __builtin_amdgcn_wmma_f32_16x16x32_bf16(false, G0, false, L1, (short)0, acc01, false, false);
    acc01 = __builtin_amdgcn_wmma_f32_16x16x32_bf16(false, A0, false, R1, (short)0, acc01, false, false);
    acc10 = __builtin_amdgcn_wmma_f32_16x16x32_bf16(false, G1, false, L0, (short)0, acc10, false, false);
    acc10 = __builtin_amdgcn_wmma_f32_16x16x32_bf16(false, A1, false, R0, (short)0, acc10, false, false);
    acc11 = __builtin_amdgcn_wmma_f32_16x16x32_bf16(false, G1, false, L1, (short)0, acc11, false, false);
    acc11 = __builtin_amdgcn_wmma_f32_16x16x32_bf16(false, A1, false, R1, (short)0, acc11, false, false);
  }

  emit_tile(acc00, nb,      ob + m,      hi, n_nodes, O, houtb, gsum);
  emit_tile(acc01, nb,      ob + 16 + m, hi, n_nodes, O, houtb, gsum);
  emit_tile(acc10, nb + 16, ob + m,      hi, n_nodes, O, houtb, gsum);
  emit_tile(acc11, nb + 16, ob + 16 + m, hi, n_nodes, O, houtb, gsum);
}

// ---- mean-pool finalize + concat manual features ----------------------------
__global__ __launch_bounds__(512) void pool_concat_kernel(const float* __restrict__ gsum,
                                                          const float* __restrict__ mf,
                                                          float* __restrict__ gvec,
                                                          int n_nodes, int H, int DM) {
  int i = blockIdx.x * blockDim.x + threadIdx.x;
  if (i < H)            gvec[i] = gsum[i] * (1.0f / (float)n_nodes);
  else if (i < H + DM)  gvec[i] = mf[i - H];
}

// ---- MLP head: relu(g @ W3^T + b3) @ W4^T + b4 ------------------------------
__global__ __launch_bounds__(128) void head_kernel(const float* __restrict__ gvec,
                                                   const float* __restrict__ W3,
                                                   const float* __restrict__ b3,
                                                   const float* __restrict__ W4,
                                                   const float* __restrict__ b4,
                                                   float* __restrict__ out,
                                                   int GD, int H3) {
  __shared__ float g[512];
  __shared__ float red[128];
  const int t = threadIdx.x;                       // t in [0, H3)
  for (int i = t; i < GD; i += blockDim.x) g[i] = gvec[i];
  __syncthreads();
  float s = b3[t];
  for (int k = 0; k < GD; ++k) s += g[k] * W3[t * GD + k];
  red[t] = fmaxf(s, 0.0f) * W4[t];
  __syncthreads();
  for (int off = 64; off > 0; off >>= 1) {
    if (t < off) red[t] += red[t + off];
    __syncthreads();
  }
  if (t == 0) out[0] = red[0] + b4[0];
}

// ---------------------------------------------------------------------------
extern "C" void kernel_launch(void* const* d_in, const int* in_sizes, int n_in,
                              void* d_out, int out_size, void* d_ws, size_t ws_size,
                              hipStream_t stream) {
  const float* x   = (const float*)d_in[0];
  const int*   ei  = (const int*)  d_in[1];   // [2,E] edge_index
  const float* mf  = (const float*)d_in[2];
  const float* W1l = (const float*)d_in[3];
  const float* b1  = (const float*)d_in[4];
  const float* W1r = (const float*)d_in[5];
  const float* W2l = (const float*)d_in[6];
  const float* b2  = (const float*)d_in[7];
  const float* W2r = (const float*)d_in[8];
  const float* W3  = (const float*)d_in[9];
  const float* b3  = (const float*)d_in[10];
  const float* W4  = (const float*)d_in[11];
  const float* b4  = (const float*)d_in[12];

  const int DIN = 128, H1 = 256, H2 = 256, DM = 16, H3 = 128;
  const int n = in_sizes[0] / DIN;
  const int E = in_sizes[1] / 2;
  const int* src = ei;
  const int* dst = ei + E;

  // ---- workspace layout (16B-aligned sections) ------------------------------
  char* w = (char*)d_ws;
  float* agg  = (float*)w;           w += (size_t)n * 256 * sizeof(float);
  float* deg  = (float*)w;           w += (((size_t)n + 63) & ~(size_t)63) * sizeof(float);
  float* gsum = (float*)w;           w += 256 * sizeof(float);
  float* gvec = (float*)w;           w += 288 * sizeof(float);
  __bf16* xb   = (__bf16*)w;         w += (size_t)n * DIN * sizeof(__bf16);
  __bf16* gb   = (__bf16*)w;         w += (size_t)n * 256 * sizeof(__bf16);
  __bf16* h1b  = (__bf16*)w;         w += (size_t)n * 256 * sizeof(__bf16);
  __bf16* wb1l = (__bf16*)w;         w += (size_t)H1 * DIN * sizeof(__bf16);
  __bf16* wb1r = (__bf16*)w;         w += (size_t)H1 * DIN * sizeof(__bf16);
  __bf16* wb2l = (__bf16*)w;         w += (size_t)H2 * H1 * sizeof(__bf16);
  __bf16* wb2r = (__bf16*)w;         w += (size_t)H2 * H1 * sizeof(__bf16);

  // ---- zero scratch (graph-capture safe; every call, for determinism) -------
  hipMemsetAsync(deg,  0, (size_t)n * sizeof(float), stream);
  hipMemsetAsync(agg,  0, (size_t)n * DIN * sizeof(float), stream);
  hipMemsetAsync(gsum, 0, 256 * sizeof(float), stream);

  // ---- one-time per-call bf16 conversions (linear, L2-resident) -------------
  cvt_bf16_kernel<<<2048, 256, 0, stream>>>(x,   xb,   (long long)n * DIN);
  cvt_bf16_kernel<<<256,  256, 0, stream>>>(W1l, wb1l, (long long)H1 * DIN);
  cvt_bf16_kernel<<<256,  256, 0, stream>>>(W1r, wb1r, (long long)H1 * DIN);
  cvt_bf16_kernel<<<256,  256, 0, stream>>>(W2l, wb2l, (long long)H2 * H1);
  cvt_bf16_kernel<<<256,  256, 0, stream>>>(W2r, wb2r, (long long)H2 * H1);

  // ---- degrees -> 1/max(deg,1) (shared by both layers) ----------------------
  degree_kernel<<<(E + 255) / 256, 256, 0, stream>>>(dst, deg, E);
  invdeg_kernel<<<(n + 255) / 256, 256, 0, stream>>>(deg, n);

  // ---- layer 1: gather bf16 x, scale+cvt, WMMA GEMM -> h1 (bf16) ------------
  scatter_add_kernel<<<8192, 256, 0, stream>>>(xb, src, dst, agg, E, DIN);
  scale_cvt_kernel<<<4096, 256, 0, stream>>>(agg, deg, gb, n, DIN);
  {
    const int waves  = ((n + 31) / 32) * (H1 / 32);
    const int blocks = (waves + 7) / 8;
    sage_wmma_kernel<<<blocks, 256, 0, stream>>>(xb, gb, wb1l, wb1r, b1,
                                                 h1b, nullptr, n, DIN, H1);
  }

  // ---- layer 2: gather bf16 h1, scale+cvt, WMMA GEMM + fused mean-pool ------
  hipMemsetAsync(agg, 0, (size_t)n * H1 * sizeof(float), stream);
  scatter_add_kernel<<<16384, 256, 0, stream>>>(h1b, src, dst, agg, E, H1);
  scale_cvt_kernel<<<8192, 256, 0, stream>>>(agg, deg, gb, n, H1);
  {
    const int waves  = ((n + 31) / 32) * (H2 / 32);
    const int blocks = (waves + 7) / 8;
    sage_wmma_kernel<<<blocks, 256, 0, stream>>>(h1b, gb, wb2l, wb2r, b2,
                                                 nullptr, gsum, n, H1, H2);
  }

  // ---- pool finalize + head -------------------------------------------------
  pool_concat_kernel<<<1, 512, 0, stream>>>(gsum, mf, gvec, n, H2, DM);
  head_kernel<<<1, 128, 0, stream>>>(gvec, W3, b3, W4, b4, (float*)d_out,
                                     H2 + DM, H3);
}
